// SOC_Mamba_V3_22230750724316
// MI455X (gfx1250) — matrix operand order repaired and checked by reading, not compiled, not run on hardware
//
#include <hip/hip_runtime.h>
#include <hip/hip_bf16.h>

// ---------------- model constants ----------------
#define B_    8
#define T_    2048
#define BT    (B_ * T_)            // 16384 tokens
#define DM    256                  // D_MODEL
#define ED_   512                  // expanded dim
#define NST   16                   // N_STATE
#define DTR   16                   // DT_RANK
#define DCONV 4
#define INF   8                    // input features
#define NL    2                    // layers
#define DBCF  (DTR + 2 * NST)      // 48
#define EPS   1e-5f
#define CHUNK 64                   // scan chunk length
#define NCH   (T_ / CHUNK)         // 32 chunks

typedef __attribute__((ext_vector_type(16))) _Float16 v16h;
typedef __attribute__((ext_vector_type(8)))  float    v8f;

union Frag16 { unsigned int u[8]; v16h h; };

// =====================================================================
// f32 -> f16 conversion (weights / feeder buffers)
// =====================================================================
__global__ void k_f32_to_f16(const float* __restrict__ in, _Float16* __restrict__ out, int n) {
    int i = blockIdx.x * blockDim.x + threadIdx.x;
    if (i < n) out[i] = (_Float16)in[i];
}

// =====================================================================
// input projection (K=8, scalar FMA) + LayerNorm.  One block per token.
// =====================================================================
__global__ void k_input_proj_ln(const float* __restrict__ x,
                                const float* __restrict__ W,   // (256,8)
                                const float* __restrict__ bias,
                                const float* __restrict__ g,
                                const float* __restrict__ bln,
                                float* __restrict__ h) {
    const int tok = blockIdx.x;          // BT
    const int d   = threadIdx.x;         // 256
    const float* xr = x + (size_t)tok * INF;
    const float* wr = W + (size_t)d * INF;
    float v = bias[d];
#pragma unroll
    for (int i = 0; i < INF; ++i) v += xr[i] * wr[i];

    __shared__ float s1[DM], s2[DM];
    s1[d] = v; s2[d] = v * v;
    __syncthreads();
    for (int off = DM / 2; off > 0; off >>= 1) {
        if (d < off) { s1[d] += s1[d + off]; s2[d] += s2[d + off]; }
        __syncthreads();
    }
    const float m   = s1[0] * (1.0f / DM);
    const float var = s2[0] * (1.0f / DM) - m * m;
    const float inv = rsqrtf(var + EPS);
    h[(size_t)tok * DM + d] = (v - m) * inv * g[d] + bln[d];
}

// =====================================================================
// RMSNorm -> f16 feeder.  One block per token.
// =====================================================================
__global__ void k_rmsnorm_f16(const float* __restrict__ h,
                              const float* __restrict__ w,
                              _Float16* __restrict__ out) {
    const int tok = blockIdx.x;
    const int d   = threadIdx.x;
    const float v = h[(size_t)tok * DM + d];
    __shared__ float s1[DM];
    s1[d] = v * v;
    __syncthreads();
    for (int off = DM / 2; off > 0; off >>= 1) {
        if (d < off) s1[d] += s1[d + off];
        __syncthreads();
    }
    const float inv = rsqrtf(s1[0] * (1.0f / DM) + EPS);
    out[(size_t)tok * DM + d] = (_Float16)(v * inv * w[d]);
}

// =====================================================================
// WMMA GEMM:  C(MxN,f32) (+)= A(MxK,f16 rm) * Bw(NxK,f16 rm)^T
// One wave32 per 16 x (16*NT) strip: A fragment loaded once per K-step
// and reused across NT v_wmma_f32_16x16x32_f16 issues (4x less A traffic,
// 4x higher WMMA density per VMEM).  N, K compile-time -> full unroll.
// Fragment layouts per CDNA5 ISA 7.12.2 (16-bit A 16x32, B 32x16).
// =====================================================================
template <int NT, int N, int K, bool ACC>
__global__ void k_wmma_gemm(const _Float16* __restrict__ A,
                            const _Float16* __restrict__ Bw,
                            float* __restrict__ C, int M) {
    const int wave    = (blockIdx.x * blockDim.x + threadIdx.x) >> 5;
    const int lane    = threadIdx.x & 31;
    const int nstrips = N / (16 * NT);
    const int tm      = wave / nstrips;
    const int tn0     = (wave % nstrips) * NT;
    if (tm * 16 >= M) return;

    const int l  = lane & 15;
    const int hi = (lane >> 4) & 1;            // lanes 16..31

    const _Float16* arow = A + (size_t)(tm * 16 + l) * K;
    const _Float16* brow[NT];
#pragma unroll
    for (int i = 0; i < NT; ++i)
        brow[i] = Bw + (size_t)((tn0 + i) * 16 + l) * K;

    const int kAoff = hi ? 8  : 0;             // A: VGPR j -> K = 2j (+8 hi), +16 for j>=4
    const int kBoff = hi ? 8  : 0;             // B: VGPR j -> K = 2j (+8 hi), +16 for j>=4

    v8f acc[NT];
#pragma unroll
    for (int i = 0; i < NT; ++i) acc[i] = (v8f){};

#pragma unroll
    for (int k0 = 0; k0 < K; k0 += 32) {
        __builtin_prefetch((const void*)(arow + k0 + 128), 0, 3);
        Frag16 a;
#pragma unroll
        for (int j = 0; j < 4; ++j) {
            a.u[j]     = *(const unsigned int*)(arow + k0 + kAoff + 2 * j);
            a.u[4 + j] = *(const unsigned int*)(arow + k0 + 16 + kAoff + 2 * j);
        }
#pragma unroll
        for (int i = 0; i < NT; ++i) {
            __builtin_prefetch((const void*)(brow[i] + k0 + 128), 0, 3);
            Frag16 b;
#pragma unroll
            for (int j = 0; j < 4; ++j) {
                b.u[j]     = *(const unsigned int*)(brow[i] + k0 + kBoff + 2 * j);
                b.u[4 + j] = *(const unsigned int*)(brow[i] + k0 + 16 + kBoff + 2 * j);
            }
            acc[i] = __builtin_amdgcn_wmma_f32_16x16x32_f16(
                         /*neg_a=*/false, a.h, /*neg_b=*/false, b.h,
                         /*c_mod=*/(short)0, acc[i],
                         /*reuse_a=*/false, /*reuse_b=*/false);
        }
    }

    // C/D layout: VGPR r -> row M0 + r (+8 for hi lanes); lane l -> col
#pragma unroll
    for (int i = 0; i < NT; ++i) {
        float* crow = C + (size_t)(tm * 16 + (hi ? 8 : 0)) * N + (tn0 + i) * 16 + l;
#pragma unroll
        for (int r = 0; r < 8; ++r) {
            float* p = crow + (size_t)r * N;
            float  v = acc[i][r];
            if (ACC) v += *p;
            *p = v;
        }
    }
}

// =====================================================================
// causal depthwise conv (width 4) + bias + SiLU.  Reads xm half of xz.
// Writes f32 (for scan) and f16 (for x_proj WMMA).
// =====================================================================
__global__ void k_conv_silu(const float* __restrict__ xz,     // (BT, 2*ED)
                            const float* __restrict__ Wc,     // (ED,1,4)
                            const float* __restrict__ bc,     // (ED)
                            float* __restrict__ xc,
                            _Float16* __restrict__ xc16) {
    const int idx = blockIdx.x * blockDim.x + threadIdx.x;    // BT*ED
    if (idx >= BT * ED_) return;
    const int e = idx % ED_;
    const int t = (idx / ED_) % T_;
    const int b = idx / (ED_ * T_);
    const size_t rowbase = (size_t)(b * T_) * (2 * ED_);

    float acc = bc[e];
#pragma unroll
    for (int j = 0; j < DCONV; ++j) {
        const int tt = t - (DCONV - 1) + j;
        if (tt >= 0)
            acc += Wc[e * DCONV + j] * xz[rowbase + (size_t)tt * (2 * ED_) + e];
    }
    const float s = acc / (1.0f + __expf(-acc));
    xc[idx]   = s;
    xc16[idx] = (_Float16)s;
}

// =====================================================================
// Chunked parallel scan (linear recurrence h_t = dA_t*h_t-1 + dBx_t).
// Pass 1: per-chunk local scan from h0=0 -> (prod dA, h_end).   512 blocks
// Pass 2: per-(b,e) sequential combine over 32 chunks -> h_init. 16 blocks
// Pass 3: replay each chunk from its true h_init, emit gated y.  512 blocks
// dt_proj (K=16) + softplus folded in; dbc[b,t,0:48] staged in LDS.
// =====================================================================
__device__ __forceinline__ void scan_preload(const float* Wdt, const float* Alog,
                                             int e, float* wdt, float* Aneg) {
#pragma unroll
    for (int n = 0; n < DTR; ++n) wdt[n] = Wdt[(size_t)e * DTR + n];
#pragma unroll
    for (int n = 0; n < NST; ++n) Aneg[n] = -__expf(Alog[(size_t)e * NST + n]);
}

__global__ void k_scan_pass1(const float* __restrict__ dbc,   // (BT,48)
                             const float* __restrict__ xc,    // (BT,ED)
                             const float* __restrict__ Wdt,   // (ED,16)
                             const float* __restrict__ bdt,   // (ED)
                             const float* __restrict__ Alog,  // (ED,16)
                             float* __restrict__ hend,        // (B,NCH,ED,NST)
                             float* __restrict__ aprod) {     // (B,NCH,ED,NST)
    const int eb  = ED_ / 256;
    int bid = blockIdx.x;
    const int ebi = bid % eb;  bid /= eb;
    const int c   = bid % NCH; bid /= NCH;
    const int b   = bid;
    const int e   = ebi * 256 + threadIdx.x;

    float wdt[DTR], Aneg[NST], hst[NST], ap[NST];
    scan_preload(Wdt, Alog, e, wdt, Aneg);
#pragma unroll
    for (int n = 0; n < NST; ++n) { hst[n] = 0.0f; ap[n] = 1.0f; }
    const float bdte = bdt[e];

    __shared__ float s[DBCF];
    for (int tt = 0; tt < CHUNK; ++tt) {
        const size_t tok = (size_t)b * T_ + c * CHUNK + tt;
        if (threadIdx.x < DBCF) s[threadIdx.x] = dbc[tok * DBCF + threadIdx.x];
        __syncthreads();

        float dt = bdte;
#pragma unroll
        for (int r = 0; r < DTR; ++r) dt += s[r] * wdt[r];
        const float delta = (dt > 20.0f) ? dt : __logf(1.0f + __expf(dt));
        const float dx    = delta * xc[tok * ED_ + e];
#pragma unroll
        for (int n = 0; n < NST; ++n) {
            const float dA = __expf(delta * Aneg[n]);
            hst[n] = dA * hst[n] + dx * s[DTR + n];
            ap[n] *= dA;
        }
        __syncthreads();
    }
    const size_t base = (((size_t)b * NCH + c) * ED_ + e) * NST;
#pragma unroll
    for (int n = 0; n < NST; ++n) { hend[base + n] = hst[n]; aprod[base + n] = ap[n]; }
}

__global__ void k_scan_pass2(float* __restrict__ hstate,       // in: hend, out: hinit
                             const float* __restrict__ aprod) {
    const int eb = ED_ / 256;
    const int b  = blockIdx.x / eb;
    const int e  = (blockIdx.x % eb) * 256 + threadIdx.x;
    float carry[NST];
#pragma unroll
    for (int n = 0; n < NST; ++n) carry[n] = 0.0f;
    for (int c = 0; c < NCH; ++c) {
        const size_t base = (((size_t)b * NCH + c) * ED_ + e) * NST;
#pragma unroll
        for (int n = 0; n < NST; ++n) {
            const float he = hstate[base + n];
            const float ap = aprod[base + n];
            hstate[base + n] = carry[n];            // becomes chunk's h_init
            carry[n] = ap * carry[n] + he;
        }
    }
}

__global__ void k_scan_pass3(const float* __restrict__ dbc,
                             const float* __restrict__ xc,
                             const float* __restrict__ xz,     // z half
                             const float* __restrict__ Wdt,
                             const float* __restrict__ bdt,
                             const float* __restrict__ Alog,
                             const float* __restrict__ Dp,
                             const float* __restrict__ hinit,  // (B,NCH,ED,NST)
                             _Float16* __restrict__ y16) {     // (BT,ED)
    const int eb  = ED_ / 256;
    int bid = blockIdx.x;
    const int ebi = bid % eb;  bid /= eb;
    const int c   = bid % NCH; bid /= NCH;
    const int b   = bid;
    const int e   = ebi * 256 + threadIdx.x;

    float wdt[DTR], Aneg[NST], hst[NST];
    scan_preload(Wdt, Alog, e, wdt, Aneg);
    const size_t sbase = (((size_t)b * NCH + c) * ED_ + e) * NST;
#pragma unroll
    for (int n = 0; n < NST; ++n) hst[n] = hinit[sbase + n];
    const float bdte = bdt[e];
    const float dpe  = Dp[e];

    __shared__ float s[DBCF];
    for (int tt = 0; tt < CHUNK; ++tt) {
        const size_t tok = (size_t)b * T_ + c * CHUNK + tt;
        if (threadIdx.x < DBCF) s[threadIdx.x] = dbc[tok * DBCF + threadIdx.x];
        __syncthreads();

        float dt = bdte;
#pragma unroll
        for (int r = 0; r < DTR; ++r) dt += s[r] * wdt[r];
        const float delta = (dt > 20.0f) ? dt : __logf(1.0f + __expf(dt));
        const float xcv   = xc[tok * ED_ + e];
        const float dx    = delta * xcv;

        float y = 0.0f;
#pragma unroll
        for (int n = 0; n < NST; ++n) {
            const float dA = __expf(delta * Aneg[n]);
            hst[n] = dA * hst[n] + dx * s[DTR + n];
            y += hst[n] * s[DTR + NST + n];
        }
        y += dpe * xcv;
        const float zv = xz[tok * (2 * ED_) + ED_ + e];
        y *= zv / (1.0f + __expf(-zv));
        y16[tok * ED_ + e] = (_Float16)y;
        __syncthreads();
    }
}

// =====================================================================
// final RMSNorm + head (2 outputs) + clip + add past.  One block/token.
// =====================================================================
__global__ void k_final_head(const float* __restrict__ h,
                             const float* __restrict__ fw,
                             const float* __restrict__ headW,  // (2,256)
                             const float* __restrict__ headb,  // (2)
                             const float* __restrict__ x,      // (BT,8)
                             float* __restrict__ out) {        // [soc(BT), soh(BT)]
    const int tok = blockIdx.x;
    const int d   = threadIdx.x;
    const float v = h[(size_t)tok * DM + d];

    __shared__ float s1[DM];
    s1[d] = v * v;
    __syncthreads();
    for (int off = DM / 2; off > 0; off >>= 1) {
        if (d < off) s1[d] += s1[d + off];
        __syncthreads();
    }
    const float hn = v * rsqrtf(s1[0] * (1.0f / DM) + EPS) * fw[d];
    __syncthreads();

    __shared__ float r0[DM], r1[DM];
    r0[d] = hn * headW[d];
    r1[d] = hn * headW[DM + d];
    __syncthreads();
    for (int off = DM / 2; off > 0; off >>= 1) {
        if (d < off) { r0[d] += r0[d + off]; r1[d] += r1[d + off]; }
        __syncthreads();
    }
    if (d == 0) {
        float d0 = r0[0] + headb[0];
        float d1 = r1[0] + headb[1];
        d0 = fminf(fmaxf(d0, -0.005f),  0.005f);
        d1 = fminf(fmaxf(d1, -0.0001f), 0.0001f);
        out[tok]      = x[(size_t)tok * INF + 4] + d0;   // past_soc
        out[BT + tok] = x[(size_t)tok * INF + 7] + d1;   // past_soh
    }
}

// =====================================================================
// host side
// =====================================================================
static inline size_t align256(size_t v) { return (v + 255) & ~(size_t)255; }

extern "C" void kernel_launch(void* const* d_in, const int* in_sizes, int n_in,
                              void* d_out, int out_size, void* d_ws, size_t ws_size,
                              hipStream_t stream) {
    const float* x       = (const float*)d_in[0];
    const float* ipW     = (const float*)d_in[1];
    const float* ipb     = (const float*)d_in[2];
    const float* ln_g    = (const float*)d_in[3];
    const float* ln_b    = (const float*)d_in[4];
    const float* inW     = (const float*)d_in[5];   // (2,1024,256)
    const float* convW   = (const float*)d_in[6];   // (2,512,1,4)
    const float* convb   = (const float*)d_in[7];   // (2,512)
    const float* xprojW  = (const float*)d_in[8];   // (2,48,512)
    const float* dtW     = (const float*)d_in[9];   // (2,512,16)
    const float* dtb     = (const float*)d_in[10];  // (2,512)
    const float* Alog    = (const float*)d_in[11];  // (2,512,16)
    const float* Dskip   = (const float*)d_in[12];  // (2,512)
    const float* outW    = (const float*)d_in[13];  // (2,256,512)
    const float* mixw    = (const float*)d_in[14];  // (2,256)
    const float* finw    = (const float*)d_in[15];
    const float* headW   = (const float*)d_in[16];
    const float* headb   = (const float*)d_in[17];
    float* out = (float*)d_out;

    // ---- workspace carve ----
    char* ws = (char*)d_ws;
    size_t cur = 0;
    auto carve = [&](size_t bytes) { char* p = ws + cur; cur += align256(bytes); return p; };
    float*    h      = (float*)   carve((size_t)BT * DM  * 4);
    _Float16* hn16   = (_Float16*)carve((size_t)BT * DM  * 2);
    float*    xz     = (float*)   carve((size_t)BT * 2 * ED_ * 4);
    float*    xc     = (float*)   carve((size_t)BT * ED_ * 4);
    _Float16* xc16   = (_Float16*)carve((size_t)BT * ED_ * 2);
    float*    dbc    = (float*)   carve((size_t)BT * DBCF * 4);
    _Float16* y16    = (_Float16*)carve((size_t)BT * ED_ * 2);
    float*    hstate = (float*)   carve((size_t)B_ * NCH * ED_ * NST * 4); // hend -> hinit
    float*    aprod  = (float*)   carve((size_t)B_ * NCH * ED_ * NST * 4);
    _Float16* inW16  = (_Float16*)carve((size_t)NL * 2 * ED_ * DM * 2);
    _Float16* xpW16  = (_Float16*)carve((size_t)NL * DBCF * ED_ * 2);
    _Float16* outW16 = (_Float16*)carve((size_t)NL * DM * ED_ * 2);
    (void)ws_size;

    // ---- weight conversion (f32 -> f16) ----
    {
        int n = NL * 2 * ED_ * DM;
        k_f32_to_f16<<<(n + 255) / 256, 256, 0, stream>>>(inW, inW16, n);
        n = NL * DBCF * ED_;
        k_f32_to_f16<<<(n + 255) / 256, 256, 0, stream>>>(xprojW, xpW16, n);
        n = NL * DM * ED_;
        k_f32_to_f16<<<(n + 255) / 256, 256, 0, stream>>>(outW, outW16, n);
    }

    // ---- input projection + LayerNorm ----
    k_input_proj_ln<<<BT, DM, 0, stream>>>(x, ipW, ipb, ln_g, ln_b, h);

    const int scan_blocks = B_ * NCH * (ED_ / 256);   // 512

    // ---- layers ----
    for (int l = 0; l < NL; ++l) {
        const float* Wdt_l  = dtW   + (size_t)l * ED_ * DTR;
        const float* bdt_l  = dtb   + (size_t)l * ED_;
        const float* Alog_l = Alog  + (size_t)l * ED_ * NST;
        const float* Dp_l   = Dskip + (size_t)l * ED_;

        // rmsnorm -> f16
        k_rmsnorm_f16<<<BT, DM, 0, stream>>>(h, mixw + (size_t)l * DM, hn16);

        // in_proj: (BT,256) x (1024,256)^T -> xz (BT,1024)   [NT=4 strips]
        {
            constexpr int NT = 4, N = 2 * ED_, K = DM;
            const int waves = (BT / 16) * (N / (16 * NT));
            k_wmma_gemm<NT, N, K, false><<<(waves * 32 + 255) / 256, 256, 0, stream>>>(
                hn16, inW16 + (size_t)l * 2 * ED_ * DM, xz, BT);
        }

        // causal conv + SiLU
        {
            const int n = BT * ED_;
            k_conv_silu<<<(n + 255) / 256, 256, 0, stream>>>(
                xz, convW + (size_t)l * ED_ * DCONV, convb + (size_t)l * ED_, xc, xc16);
        }

        // x_proj: (BT,512) x (48,512)^T -> dbc (BT,48)       [NT=3 strip]
        {
            constexpr int NT = 3, N = DBCF, K = ED_;
            const int waves = (BT / 16) * (N / (16 * NT));
            k_wmma_gemm<NT, N, K, false><<<(waves * 32 + 255) / 256, 256, 0, stream>>>(
                xc16, xpW16 + (size_t)l * DBCF * ED_, dbc, BT);
        }

        // chunked parallel scan
        k_scan_pass1<<<scan_blocks, 256, 0, stream>>>(
            dbc, xc, Wdt_l, bdt_l, Alog_l, hstate, aprod);
        k_scan_pass2<<<B_ * (ED_ / 256), 256, 0, stream>>>(hstate, aprod);
        k_scan_pass3<<<scan_blocks, 256, 0, stream>>>(
            dbc, xc, xz, Wdt_l, bdt_l, Alog_l, Dp_l, hstate, y16);

        // out_proj + residual: h += (BT,512) x (256,512)^T   [NT=4 strips]
        {
            constexpr int NT = 4, N = DM, K = ED_;
            const int waves = (BT / 16) * (N / (16 * NT));
            k_wmma_gemm<NT, N, K, true><<<(waves * 32 + 255) / 256, 256, 0, stream>>>(
                y16, outW16 + (size_t)l * DM * ED_, h, BT);
        }
    }

    // ---- final norm + head ----
    k_final_head<<<BT, DM, 0, stream>>>(h, finw, headW, headb, x, out);
    (void)in_sizes; (void)n_in; (void)out_size;
}
